// DCFMBlock_35081292873776
// MI455X (gfx1250) — matrix operand order repaired and checked by reading, not compiled, not run on hardware
//
#include <hip/hip_runtime.h>
#include <hip/hip_bf16.h>
#include <stdint.h>

// Problem constants (from reference)
#define B_  8
#define S_  3
#define H_  48
#define W_  48
#define DM_ 96
#define DE_ 192
#define N_  16
#define R_  6
#define G_  4
#define L_  (H_*W_)     // 2304
#define L2_ (2*L_)      // 4608
#define P_  38          // R + 2N
#define PP_ 48          // P padded to 3 WMMA col-tiles

typedef __attribute__((ext_vector_type(16))) __bf16 v16bf;
typedef __attribute__((ext_vector_type(8)))  float  v8f;

struct __align__(16) U4 { unsigned int x, y, z, w; };
union FragBF { v16bf v; U4 q[2]; };

__device__ __forceinline__ unsigned short f2bf(float f) {
  unsigned int u = __float_as_uint(f);
  u += 0x7FFFu + ((u >> 16) & 1u);      // round-to-nearest-even
  return (unsigned short)(u >> 16);
}

// ---------------------------------------------------------------------------
// K0: weight prep — bf16 conversions (+ zero-pad x_proj rows 38->48), A = -exp(A_logs)
// ---------------------------------------------------------------------------
__global__ void k0_prep(const float* __restrict__ in_proj_w,
                        const float* __restrict__ x_proj_w,
                        const float* __restrict__ out_w0,
                        const float* __restrict__ out_w1,
                        const float* __restrict__ A_logs,
                        unsigned short* __restrict__ wInBf,
                        unsigned short* __restrict__ wXpBf,
                        unsigned short* __restrict__ wOutBf,
                        float* __restrict__ Aneg)
{
  int idx = blockIdx.x * blockDim.x + threadIdx.x;
  const int c1 = DE_ * DM_;       // 18432
  const int c2 = G_ * PP_ * DE_;  // 36864
  const int c3 = 2 * DM_ * DE_;   // 36864
  const int c4 = G_ * DE_ * N_;   // 12288
  if (idx < c1) { wInBf[idx] = f2bf(in_proj_w[idx]); return; }
  idx -= c1;
  if (idx < c2) {
    int c = idx % DE_; int p = (idx / DE_) % PP_; int g = idx / (DE_ * PP_);
    float v = (p < P_) ? x_proj_w[(g * P_ + p) * DE_ + c] : 0.0f;
    wXpBf[idx] = f2bf(v); return;
  }
  idx -= c2;
  if (idx < c3) {
    const float* w = (idx < DM_ * DE_) ? out_w0 : out_w1;
    wOutBf[idx] = f2bf(w[idx % (DM_ * DE_)]); return;
  }
  idx -= c3;
  if (idx < c4) { Aneg[idx] = -__expf(A_logs[idx]); }
}

// ---------------------------------------------------------------------------
// K1: fused LayerNorm(DM) + in_proj GEMM (96 -> 192) via bf16 WMMA, K=96 = 3x32
// Block: 256 thr (8 waves) handles one 16-row tile; wave w covers col tiles w, w+8.
// ---------------------------------------------------------------------------
__global__ __launch_bounds__(256)
void k1_ln_inproj(const float* __restrict__ x,
                  const float* __restrict__ g_ln, const float* __restrict__ b_ln,
                  const unsigned short* __restrict__ wInBf,
                  float* __restrict__ h_out)
{
  __shared__ float xs[16 * DM_];
  __shared__ __align__(16) unsigned short abf[16 * DM_];
  const int rt = blockIdx.x;
  const int tid = threadIdx.x;
  for (int e = tid; e < 16 * DM_; e += 256)
    xs[e] = x[(size_t)rt * 16 * DM_ + e];
  __syncthreads();
  if (tid < 16) {
    const float* row = &xs[tid * DM_];
    float m = 0.f;
    for (int c = 0; c < DM_; ++c) m += row[c];
    m *= (1.0f / DM_);
    float v = 0.f;
    for (int c = 0; c < DM_; ++c) { float d = row[c] - m; v += d * d; }
    float rs = rsqrtf(v * (1.0f / DM_) + 1e-5f);
    for (int c = 0; c < DM_; ++c)
      abf[tid * DM_ + c] = f2bf((row[c] - m) * rs * g_ln[c] + b_ln[c]);
  }
  __syncthreads();
  const int wave = tid >> 5, lane = tid & 31, lo = lane & 15, hi = lane >> 4;
  for (int ct = wave; ct < DE_ / 16; ct += 8) {        // wave-uniform: EXEC all ones
    v8f acc = {};
#pragma unroll
    for (int ks = 0; ks < 3; ++ks) {
      const int k0 = ks * 32;
      FragBF a, bq;
      a.q[0] = *reinterpret_cast<const U4*>(&abf[lo * DM_ + k0 + hi * 16]);
      a.q[1] = *reinterpret_cast<const U4*>(&abf[lo * DM_ + k0 + hi * 16 + 8]);
      const unsigned short* bp = &wInBf[(ct * 16 + lo) * DM_ + k0 + hi * 16];
      bq.q[0] = reinterpret_cast<const U4*>(bp)[0];
      bq.q[1] = reinterpret_cast<const U4*>(bp)[1];
      acc = __builtin_amdgcn_wmma_f32_16x16x32_bf16(false, a.v, false, bq.v,
                                                    (short)0, acc, false, false);
    }
#pragma unroll
    for (int r = 0; r < 8; ++r) {
      const int row = hi * 8 + r;
      h_out[((size_t)rt * 16 + row) * DE_ + ct * 16 + lo] = acc[r];
    }
  }
}

// ---------------------------------------------------------------------------
// K2: depthwise 3x3 conv (SAME) + bias + SiLU; writes u in fp32 (B,2,DE,L2) for
// the scan and bf16 transposed (B,2,L2,DE) for the x_proj WMMA GEMM.
// s=0 -> (i=0, t=l), s=1 -> (i=1, t=l), s=2 -> (i=0 and 1, t=L+l).
// ---------------------------------------------------------------------------
__global__ __launch_bounds__(DE_)
void k2_conv_silu(const float* __restrict__ h,
                  const float* __restrict__ cw, const float* __restrict__ cb,
                  float* __restrict__ u32, unsigned short* __restrict__ ubf)
{
  const int c = threadIdx.x;                 // channel e in [0,192)
  const int y = blockIdx.x % H_;
  const int s = (blockIdx.x / H_) % S_;
  const int b = blockIdx.x / (H_ * S_);
  const int ch = s * DE_ + c;
  float wv[9];
#pragma unroll
  for (int k = 0; k < 9; ++k) wv[k] = cw[ch * 9 + k];
  const float bias = cb[ch];
  const size_t hbase = (size_t)(b * S_ + s) * L_ * DE_;
  for (int xx0 = 0; xx0 < W_; ++xx0) {
    float acc = bias;
#pragma unroll
    for (int dy = -1; dy <= 1; ++dy) {
      const int yy = y + dy;
      if (yy < 0 || yy >= H_) continue;
#pragma unroll
      for (int dx = -1; dx <= 1; ++dx) {
        const int xx = xx0 + dx;
        if (xx < 0 || xx >= W_) continue;
        acc += h[hbase + (size_t)(yy * W_ + xx) * DE_ + c] * wv[(dy + 1) * 3 + (dx + 1)];
      }
    }
    const float v = acc / (1.f + __expf(-acc));   // SiLU
    const int l = y * W_ + xx0;
    const unsigned short vb = f2bf(v);
    if (s < 2) {
      u32[((size_t)(b * 2 + s) * DE_ + c) * L2_ + l] = v;
      ubf[((size_t)(b * 2 + s) * L2_ + l) * DE_ + c] = vb;
    } else {
#pragma unroll
      for (int i = 0; i < 2; ++i) {
        u32[((size_t)(b * 2 + i) * DE_ + c) * L2_ + L_ + l] = v;
        ubf[((size_t)(b * 2 + i) * L2_ + L_ + l) * DE_ + c] = vb;
      }
    }
  }
}

// ---------------------------------------------------------------------------
// K3: x_proj GEMM: zT[b,g,t,p] = sum_c u[b,g%2,c,t] * x_proj_w[g,p,c]
// The 16x192 bf16 A-tile (one contiguous 6144 B chunk of ubf) is staged into
// LDS ONCE per block with GLOBAL_LOAD_ASYNC_TO_LDS_B128 (ASYNCcnt), then all
// 3 waves feed their WMMAs from LDS — 3x less global activation traffic.
// Block: 96 thr (3 waves), one 16-row tile, wave = col tile.
// ---------------------------------------------------------------------------
__global__ __launch_bounds__(96)
void k3_xproj(const unsigned short* __restrict__ ubf,
              const unsigned short* __restrict__ wXpBf,
              float* __restrict__ zT)
{
  __shared__ __align__(16) unsigned short atile[16 * DE_];   // 6144 B
  const int mt = blockIdx.x % (L2_ / 16);
  const int g  = (blockIdx.x / (L2_ / 16)) % G_;
  const int b  = blockIdx.x / (G_ * (L2_ / 16));
  const int i  = g & 1;
  const int tid = threadIdx.x;

  // Async stage: 96 threads x 4 x b128 = 6144 B, contiguous copy.
  {
    const unsigned short* asrc = &ubf[((size_t)(b * 2 + i) * L2_ + mt * 16) * DE_];
    unsigned int ldsOff = (unsigned int)(uintptr_t)(&atile[0]) + (unsigned int)tid * 16u;
    const char* gsrc = (const char*)asrc + tid * 16;
#pragma unroll
    for (int k = 0; k < 4; ++k) {
      asm volatile("global_load_async_to_lds_b128 %0, %1, off"
                   :: "v"(ldsOff + (unsigned int)(k * 1536)),
                      "v"(gsrc + k * 1536)
                   : "memory");
    }
    asm volatile("s_wait_asynccnt 0x0" ::: "memory");
  }
  __syncthreads();

  const int ct = tid >> 5, lane = tid & 31, lo = lane & 15, hi = lane >> 4;
  const unsigned short* brow = &wXpBf[((size_t)g * PP_ + ct * 16 + lo) * DE_];
  v8f acc = {};
#pragma unroll
  for (int ks = 0; ks < 6; ++ks) {
    const int k0 = ks * 32;
    FragBF a, bq;
    a.q[0] = *reinterpret_cast<const U4*>(&atile[lo * DE_ + k0 + hi * 16]);
    a.q[1] = *reinterpret_cast<const U4*>(&atile[lo * DE_ + k0 + hi * 16 + 8]);
    bq.q[0] = reinterpret_cast<const U4*>(brow + k0 + hi * 16)[0];
    bq.q[1] = reinterpret_cast<const U4*>(brow + k0 + hi * 16)[1];
    acc = __builtin_amdgcn_wmma_f32_16x16x32_bf16(false, a.v, false, bq.v,
                                                  (short)0, acc, false, false);
  }
  const int p = ct * 16 + lo;
  if (p < P_) {
#pragma unroll
    for (int r = 0; r < 8; ++r) {
      const int t = mt * 16 + hi * 8 + r;
      zT[((size_t)(b * G_ + g) * L2_ + t) * PP_ + p] = acc[r];
    }
  }
}

// ---------------------------------------------------------------------------
// K4: selective scan. 16-lane group per (b, i, c, dir); lane n owns state n.
// dt computed on the fly from zT rows 0..5 (+softplus). Only the L steps that
// feed y (j=0 half) are executed: fwd = first L, bwd = first L of reversed seq.
// y partial reduced with a 4-step shuffle butterfly; lane0 writes + D*u term.
// Prefetch (global_prefetch_b8) pulls the zT row 16 steps ahead into cache.
// ---------------------------------------------------------------------------
__global__ __launch_bounds__(256)
void k4_scan(const float* __restrict__ zT,
             const float* __restrict__ u32,
             const float* __restrict__ dtw_all,
             const float* __restrict__ dtb_all,
             const float* __restrict__ Aneg,
             const float* __restrict__ Ds,
             float* __restrict__ ysF, float* __restrict__ ysB)
{
  const int tid = blockIdx.x * blockDim.x + threadIdx.x;
  const int n   = tid & 15;
  const int grp = tid >> 4;               // 0..6143
  const int dir = grp & 1;
  const int rem = grp >> 1;               // 0..3071
  const int c = rem % DE_;
  const int i = (rem / DE_) & 1;
  const int b = rem / (DE_ * 2);
  const int g = i + 2 * dir;
  float dw[R_];
#pragma unroll
  for (int r = 0; r < R_; ++r) dw[r] = dtw_all[((size_t)g * DE_ + c) * R_ + r];
  const float dtb = dtb_all[g * DE_ + c];
  const float An  = Aneg[((size_t)g * DE_ + c) * N_ + n];
  const float Dsc = Ds[g * DE_ + c];
  const float* zbase = zT + (size_t)(b * G_ + g) * L2_ * PP_;
  const float* ub    = u32 + ((size_t)(b * 2 + i) * DE_ + c) * L2_;
  float* yout = (dir ? ysB : ysF) + ((size_t)(b * 2 + i) * DE_ + c) * L_;
  float hst = 0.f;
  for (int j = 0; j < L_; ++j) {
    const int t = dir ? (L2_ - 1 - j) : j;
    const float* zrow = zbase + (size_t)t * PP_;
    // prefetch the row 16 steps ahead (clamped in-bounds; speculative RT hint)
    {
      int tp = t + (dir ? -16 : 16);
      tp = tp < 0 ? 0 : (tp >= L2_ ? L2_ - 1 : tp);
      __builtin_prefetch(zbase + (size_t)tp * PP_, 0, 0);
    }
    float dtr = dtb;
#pragma unroll
    for (int r = 0; r < R_; ++r) dtr += dw[r] * zrow[r];
    const float dt = (dtr > 20.f) ? dtr : __logf(1.f + __expf(dtr));   // softplus
    const float uv = ub[t];
    hst = __expf(dt * An) * hst + (dt * uv) * zrow[R_ + n];            // B_n
    float part = hst * zrow[R_ + N_ + n];                               // C_n
    part += __shfl_xor(part, 1);
    part += __shfl_xor(part, 2);
    part += __shfl_xor(part, 4);
    part += __shfl_xor(part, 8);
    if (n == 0) {
      const int l = dir ? (L_ - 1 - j) : j;
      yout[l] = part + Dsc * uv;
    }
  }
}

// ---------------------------------------------------------------------------
// K5: y = ysF+ysB (transpose load), LayerNorm(DE), out_proj GEMM (192 -> 96)
// via bf16 WMMA, + SiLU; writes xs = silu(o) straight into d_out.
// Block: 192 thr (6 waves), one 16-row tile, wave = col tile (6 of DM=96).
// ---------------------------------------------------------------------------
__global__ __launch_bounds__(192)
void k5_out(const float* __restrict__ ysF, const float* __restrict__ ysB,
            const float* __restrict__ g_ln, const float* __restrict__ b_ln,
            const unsigned short* __restrict__ wOutBf,
            float* __restrict__ out)
{
  __shared__ float ys[16 * DE_];
  __shared__ __align__(16) unsigned short abf[16 * DE_];
  const int lt = blockIdx.x % (L_ / 16);
  const int i  = (blockIdx.x / (L_ / 16)) % 2;
  const int b  = blockIdx.x / (2 * (L_ / 16));
  const int tid = threadIdx.x;
  const size_t base = (size_t)(b * 2 + i) * DE_ * L_;
  for (int e = tid; e < 16 * DE_; e += 192) {
    const int cc = e >> 4, lo2 = e & 15;
    const size_t gi = base + (size_t)cc * L_ + lt * 16 + lo2;
    ys[lo2 * DE_ + cc] = ysF[gi] + ysB[gi];
  }
  __syncthreads();
  if (tid < 16) {
    const float* row = &ys[tid * DE_];
    float m = 0.f;
    for (int c2 = 0; c2 < DE_; ++c2) m += row[c2];
    m *= (1.0f / DE_);
    float v = 0.f;
    for (int c2 = 0; c2 < DE_; ++c2) { float d = row[c2] - m; v += d * d; }
    const float rs = rsqrtf(v * (1.0f / DE_) + 1e-5f);
    for (int c2 = 0; c2 < DE_; ++c2)
      abf[tid * DE_ + c2] = f2bf((row[c2] - m) * rs * g_ln[c2] + b_ln[c2]);
  }
  __syncthreads();
  const int wave = tid >> 5, lane = tid & 31, lo = lane & 15, hi = lane >> 4;
  const int ct = wave;                                 // 0..5
  const unsigned short* wb = wOutBf + (size_t)i * DM_ * DE_;
  v8f acc = {};
#pragma unroll
  for (int ks = 0; ks < 6; ++ks) {
    const int k0 = ks * 32;
    FragBF a, bq;
    a.q[0] = *reinterpret_cast<const U4*>(&abf[lo * DE_ + k0 + hi * 16]);
    a.q[1] = *reinterpret_cast<const U4*>(&abf[lo * DE_ + k0 + hi * 16 + 8]);
    const unsigned short* bp = wb + (size_t)(ct * 16 + lo) * DE_ + k0 + hi * 16;
    bq.q[0] = reinterpret_cast<const U4*>(bp)[0];
    bq.q[1] = reinterpret_cast<const U4*>(bp)[1];
    acc = __builtin_amdgcn_wmma_f32_16x16x32_bf16(false, a.v, false, bq.v,
                                                  (short)0, acc, false, false);
  }
#pragma unroll
  for (int r = 0; r < 8; ++r) {
    const int row = hi * 8 + r;
    const float val = acc[r];
    const float o = val / (1.f + __expf(-val));        // SiLU (cab input xs)
    out[((size_t)(b * 2 + i) * L_ + lt * 16 + row) * DM_ + ct * 16 + lo] = o;
  }
}

// ---------------------------------------------------------------------------
// K6a: CAB gate — per (b,i): avg & max over L pixels, tiny 96x96 GEMMs, gate.
// ---------------------------------------------------------------------------
__global__ __launch_bounds__(DM_)
void k6_cab_gate(const float* __restrict__ out,
                 const float* __restrict__ w0, const float* __restrict__ b0,
                 const float* __restrict__ w1, const float* __restrict__ b1,
                 float* __restrict__ ca)
{
  __shared__ float sav[DM_], smx[DM_];
  const int i = blockIdx.x & 1;
  const int b = blockIdx.x >> 1;
  const int m = threadIdx.x;
  const float* base = out + (size_t)(b * 2 + i) * L_ * DM_;
  float s = 0.f, mx = -3.4e38f;
  for (int l = 0; l < L_; ++l) {
    const float v = base[(size_t)l * DM_ + m];
    s += v; mx = fmaxf(mx, v);
  }
  sav[m] = s * (1.0f / L_);
  smx[m] = mx;
  __syncthreads();
  const float* w  = i ? w1 : w0;
  const float* bb = i ? b1 : b0;
  float ga = bb[m], gm = bb[m];
  for (int k = 0; k < DM_; ++k) {
    const float wk = w[m * DM_ + k];
    ga += wk * sav[k];
    gm += wk * smx[k];
  }
  const float gsum = fmaxf(ga, 0.f) + fmaxf(gm, 0.f);
  ca[(b * 2 + i) * DM_ + m] = 1.f + 1.f / (1.f + __expf(-gsum));
}

// K6b: apply channel attention in place on d_out.
__global__ __launch_bounds__(256)
void k6_scale(float* __restrict__ out, const float* __restrict__ ca, int total)
{
  for (int idx = blockIdx.x * blockDim.x + threadIdx.x; idx < total;
       idx += gridDim.x * blockDim.x) {
    const int m  = idx % DM_;
    const int bi = idx / (L_ * DM_);
    out[idx] *= ca[bi * DM_ + m];
  }
}

// ---------------------------------------------------------------------------
extern "C" void kernel_launch(void* const* d_in, const int* in_sizes, int n_in,
                              void* d_out, int out_size, void* d_ws, size_t ws_size,
                              hipStream_t stream)
{
  const float* x         = (const float*)d_in[0];
  const float* in_g      = (const float*)d_in[1];
  const float* in_b      = (const float*)d_in[2];
  const float* in_proj_w = (const float*)d_in[3];
  const float* conv_w    = (const float*)d_in[4];
  const float* conv_b    = (const float*)d_in[5];
  const float* x_proj_w  = (const float*)d_in[6];
  const float* dt_w      = (const float*)d_in[7];
  const float* dt_b      = (const float*)d_in[8];
  const float* A_logs    = (const float*)d_in[9];
  const float* Ds        = (const float*)d_in[10];
  const float* on_g      = (const float*)d_in[11];
  const float* on_b      = (const float*)d_in[12];
  const float* ow0       = (const float*)d_in[13];
  const float* ow1       = (const float*)d_in[14];
  const float* cw0       = (const float*)d_in[15];
  const float* cb0       = (const float*)d_in[16];
  const float* cw1       = (const float*)d_in[17];
  const float* cb1       = (const float*)d_in[18];
  float* out = (float*)d_out;

  char* ws = (char*)d_ws;
  size_t off = 0;
  auto alloc = [&](size_t bytes) {
    char* p = ws + off;
    off = (off + bytes + 255) & ~(size_t)255;
    return p;
  };
  unsigned short* wInBf  = (unsigned short*)alloc((size_t)DE_ * DM_ * 2);
  unsigned short* wXpBf  = (unsigned short*)alloc((size_t)G_ * PP_ * DE_ * 2);
  unsigned short* wOutBf = (unsigned short*)alloc((size_t)2 * DM_ * DE_ * 2);
  float* Aneg = (float*)alloc((size_t)G_ * DE_ * N_ * 4);
  float* hbuf = (float*)alloc((size_t)B_ * S_ * L_ * DE_ * 4);   // 42.5 MB
  float* u32  = (float*)alloc((size_t)B_ * 2 * DE_ * L2_ * 4);   // 56.6 MB
  unsigned short* ubf = (unsigned short*)alloc((size_t)B_ * 2 * L2_ * DE_ * 2); // 28.3 MB
  float* zT  = (float*)alloc((size_t)B_ * G_ * L2_ * PP_ * 4);   // 28.3 MB
  float* ysB = (float*)alloc((size_t)B_ * 2 * DE_ * L_ * 4);     // 28.3 MB
  float* ca  = (float*)alloc((size_t)B_ * 2 * DM_ * 4);
  float* ysF = hbuf;   // reuse: h is dead after the conv, ysF fits inside

  k0_prep<<<dim3(457), dim3(256), 0, stream>>>(in_proj_w, x_proj_w, ow0, ow1,
                                               A_logs, wInBf, wXpBf, wOutBf, Aneg);
  k1_ln_inproj<<<dim3(B_ * S_ * L_ / 16), dim3(256), 0, stream>>>(x, in_g, in_b,
                                                                  wInBf, hbuf);
  k2_conv_silu<<<dim3(B_ * S_ * H_), dim3(DE_), 0, stream>>>(hbuf, conv_w, conv_b,
                                                             u32, ubf);
  k3_xproj<<<dim3(B_ * G_ * (L2_ / 16)), dim3(96), 0, stream>>>(ubf, wXpBf, zT);
  k4_scan<<<dim3((B_ * 2 * DE_ * 2 * 16) / 256), dim3(256), 0, stream>>>(
      zT, u32, dt_w, dt_b, Aneg, Ds, ysF, ysB);
  k5_out<<<dim3(B_ * 2 * (L_ / 16)), dim3(192), 0, stream>>>(ysF, ysB, on_g, on_b,
                                                             wOutBf, out);
  k6_cab_gate<<<dim3(B_ * 2), dim3(DM_), 0, stream>>>(out, cw0, cb0, cw1, cb1, ca);
  k6_scale<<<dim3(1024), dim3(256), 0, stream>>>(out, ca, B_ * 2 * L_ * DM_);

  (void)in_sizes; (void)n_in; (void)out_size; (void)ws_size;
}
